// AttentionModel_39848706572583
// MI455X (gfx1250) — matrix operand order repaired and checked by reading, not compiled, not run on hardware
//
#include <hip/hip_runtime.h>

#define BB 4
#define SS 4096
#define EE 512

typedef __attribute__((ext_vector_type(16))) __bf16 v16bf;
typedef __attribute__((ext_vector_type(8)))  __bf16 v8bf;
typedef __attribute__((ext_vector_type(8)))  float  v8f;
typedef __attribute__((ext_vector_type(4)))  float  v4f;
typedef __attribute__((ext_vector_type(4)))  unsigned int v4u;
typedef __attribute__((ext_vector_type(8)))  int    v8i;
typedef __attribute__((ext_vector_type(4)))  int    v4i;

// ---- small helpers -------------------------------------------------------

__device__ __forceinline__ v8f zero8() {
    v8f r;
#pragma unroll
    for (int i = 0; i < 8; ++i) r[i] = 0.0f;
    return r;
}

// load 8 contiguous f32 (32B aligned) and round to bf16
__device__ __forceinline__ v8bf cvt8(const float* __restrict__ p) {
    const v4f a = *(const v4f*)p;
    const v4f b = *(const v4f*)(p + 4);
    v8bf r;
#pragma unroll
    for (int i = 0; i < 4; ++i) { r[i] = (__bf16)a[i]; r[4 + i] = (__bf16)b[i]; }
    return r;
}

__device__ __forceinline__ v16bf join16(v8bf lo, v8bf hi) {
    v16bf r;
#pragma unroll
    for (int i = 0; i < 8; ++i) { r[i] = lo[i]; r[8 + i] = hi[i]; }
    return r;
}

// ---- Tensor Data Mover: 2D bf16 tile  global -> LDS ----------------------
// D# packing per CDNA5 ISA.
//   pad_interval code 7 = every 256 DWORDs (one 512-elem bf16 row),
//   pad_amount   code 3 = +4 DWORDs (= +8 bf16) -> LDS row stride 520 elems.
__device__ __forceinline__ void tdm_load_2d_bf16(
    unsigned lds_byte_addr, const void* gptr,
    unsigned tensor_d0, unsigned tensor_d1,      // tensor extent (elems, rows)
    unsigned tile_d0,   unsigned tile_d1,        // tile extent   (elems, rows)
    unsigned row_stride_elems,                   // tensor dim0 stride (elems)
    bool pad, unsigned pad_interval_code, unsigned pad_amount_code)
{
    const unsigned long long ga = (unsigned long long)gptr;
    v4u g0;
    g0[0] = 1u;                                          // count=1, user mode
    g0[1] = lds_byte_addr;                               // lds_addr[63:32]
    g0[2] = (unsigned)(ga & 0xFFFFFFFFu);                // global_addr lo
    g0[3] = (unsigned)((ga >> 32) & 0x01FFFFFFu)         // global_addr[56:32]
          | (2u << 30);                                  // type = 2 ("image")
    v8i g1;
    g1[0] = (int)((1u << 16)                             // data_size = 2 bytes
          | (pad ? (1u << 20) : 0u)                      // pad_enable
          | (pad_interval_code << 22)
          | (pad_amount_code << 25));
    g1[1] = (int)((tensor_d0 & 0xFFFFu) << 16);          // tensor_dim0[15:0]
    g1[2] = (int)((tensor_d0 >> 16) | ((tensor_d1 & 0xFFFFu) << 16));
    g1[3] = (int)((tensor_d1 >> 16) | ((tile_d0 & 0xFFFFu) << 16));
    g1[4] = (int)(tile_d1 & 0xFFFFu);                    // tile_dim1 (tile_dim2=0)
    g1[5] = (int)row_stride_elems;                       // dim0_stride[31:0]
    g1[6] = 0;                                           // stride hi, dim1_stride lo
    g1[7] = 0;
    const v4i gz = {0, 0, 0, 0};
#if __clang_major__ >= 23
    const v8i gz8 = {0, 0, 0, 0, 0, 0, 0, 0};
    __builtin_amdgcn_tensor_load_to_lds(g0, g1, gz, gz, gz8, 0);
#else
    __builtin_amdgcn_tensor_load_to_lds(g0, g1, gz, gz, 0);
#endif
}

// ==========================================================================
// Kernel 1: Q/K/V projections.  Y = X @ W^T + b   (per flat row of [B*S, E])
//   z==0 -> Qb row-major bf16, z==1 -> Kb row-major bf16,
//   z==2 -> Vt[b][d][s] (transposed) bf16 so P@V B-fragments are contiguous.
// ==========================================================================
__global__ __launch_bounds__(128) void proj_kernel(
    const float* __restrict__ q,  const float* __restrict__ k,  const float* __restrict__ v,
    const float* __restrict__ Wq, const float* __restrict__ bq,
    const float* __restrict__ Wk, const float* __restrict__ bk,
    const float* __restrict__ Wv, const float* __restrict__ bv,
    __bf16* __restrict__ Qb, __bf16* __restrict__ Kb, __bf16* __restrict__ Vt)
{
    const int z = blockIdx.z;
    const float* X    = (z == 0) ? q  : (z == 1) ? k  : v;
    const float* W    = (z == 0) ? Wq : (z == 1) ? Wk : Wv;
    const float* bias = (z == 0) ? bq : (z == 1) ? bk : bv;

    const int lane = threadIdx.x & 31;
    const int w    = threadIdx.x >> 5;
    const int half = lane >> 4;
    const int l16  = lane & 15;

    const int rbase = blockIdx.y * 64 + w * 16;  // flat row base in [0, B*S)
    const int cbase = blockIdx.x * 16;           // output feature base

    const float* xrow = X + (size_t)(rbase + l16) * EE;
    const float* wrow = W + (size_t)(cbase + l16) * EE;

    v8f c = zero8();
#pragma unroll 4
    for (int e = 0; e < EE; e += 32) {
        v16bf a = join16(cvt8(xrow + e + half * 16),
                         cvt8(xrow + e + half * 16 + 8));
        v16bf b = join16(cvt8(wrow + e + half * 16),
                         cvt8(wrow + e + half * 16 + 8));
        c = __builtin_amdgcn_wmma_f32_16x16x32_bf16(false, a, false, b,
                                                    (short)0, c, false, false);
    }

    const float bias_c = bias[cbase + l16];

#pragma unroll
    for (int r = 0; r < 8; ++r) {
        const float val = c[r] + bias_c;
        const int   row = rbase + half * 8 + r;
        if (z == 2) {
            const int bi = row >> 12;
            const int s  = row & (SS - 1);
            Vt[(size_t)bi * EE * SS + (size_t)(cbase + l16) * SS + s] = (__bf16)val;
        } else {
            __bf16* Y = (z == 0) ? Qb : Kb;
            Y[(size_t)row * EE + cbase + l16] = (__bf16)val;
        }
    }
}

// ==========================================================================
// Kernel 2: fused flash attention (no 1/sqrt(d) scale, per reference).
//   grid = (E/128 d-chunks, S/64 q-tiles, B); block = 128 threads = 4 waves.
//   Q tile TDM-loaded to LDS once; K tiles TDM double-buffered in LDS.
// ==========================================================================
#define QTILE   64
#define DCHUNK  128
#define KTILE   32
#define KSTRIDE (EE + 8)   // LDS row stride in bf16 elems (= TDM pad codes 7/3)
#define PSTRIDE 40         // LDS P row stride in bf16 elems

__global__ __launch_bounds__(128) void flash_kernel(
    const __bf16* __restrict__ Qb, const __bf16* __restrict__ Kb,
    const __bf16* __restrict__ Vt, float* __restrict__ out)
{
    __shared__ __align__(16) __bf16 ldsQ[QTILE * KSTRIDE];
    __shared__ __align__(16) __bf16 ldsK[2][KTILE * KSTRIDE];
    __shared__ __align__(16) __bf16 ldsP[4][16 * PSTRIDE];

    const int lane = threadIdx.x & 31;
    const int w    = threadIdx.x >> 5;
    const int half = lane >> 4;
    const int l16  = lane & 15;

    const int b     = blockIdx.z;
    const int qbase = blockIdx.y * QTILE;
    const int dbase = blockIdx.x * DCHUNK;

    const __bf16* qtile_g = Qb + ((size_t)b * SS + qbase) * EE;
    const __bf16* kmat    = Kb + (size_t)b * SS * EE;
    const __bf16* vmat    = Vt + (size_t)b * EE * SS;
    const __bf16* myQrow  = &ldsQ[(w * 16 + l16) * KSTRIDE];
    __bf16*       myP     = &ldsP[w][0];

    // ---- prologue: TDM-load Q tile (once) and K tile 0 (buffer 0) --------
    if (w == 0) {
        tdm_load_2d_bf16((unsigned)(size_t)(void*)&ldsQ[0], qtile_g,
                         EE, QTILE, EE, QTILE, EE, true, 7u, 3u);
        tdm_load_2d_bf16((unsigned)(size_t)(void*)&ldsK[0][0], kmat,
                         EE, SS, EE, KTILE, EE, true, 7u, 3u);
    }

    v8f acc[8];
#pragma unroll
    for (int n = 0; n < 8; ++n) acc[n] = zero8();
    float mrow[8], lrow[8];
#pragma unroll
    for (int r = 0; r < 8; ++r) { mrow[r] = -__builtin_inff(); lrow[r] = 0.0f; }

    const int NT = SS / KTILE;  // 128 key tiles
    for (int i = 0; i < NT; ++i) {
        const int kb  = i * KTILE;
        const int buf = i & 1;

        if (w == 0) {
            if (i + 1 < NT) {
                tdm_load_2d_bf16((unsigned)(size_t)(void*)&ldsK[buf ^ 1][0],
                                 kmat + (size_t)(kb + KTILE) * EE,
                                 EE, SS, EE, KTILE, EE, true, 7u, 3u);
                __builtin_amdgcn_s_wait_tensorcnt(1);   // tile i has landed
            } else {
                __builtin_amdgcn_s_wait_tensorcnt(0);
            }
        }
        __syncthreads();

        const __bf16* kt = &ldsK[buf][0];

        // ---- scores S[16][32] = Q_tile @ K_tile^T ------------------------
        v8f c0 = zero8(), c1 = zero8();
        for (int e = 0; e < EE; e += 32) {
            const v16bf a = join16(*(const v8bf*)(myQrow + e + half * 16),
                                   *(const v8bf*)(myQrow + e + half * 16 + 8));
            const __bf16* kr0 = kt + l16 * KSTRIDE + e + half * 16;
            const __bf16* kr1 = kt + (16 + l16) * KSTRIDE + e + half * 16;
            const v16bf bf0 = join16(*(const v8bf*)kr0, *(const v8bf*)(kr0 + 8));
            const v16bf bf1 = join16(*(const v8bf*)kr1, *(const v8bf*)(kr1 + 8));
            c0 = __builtin_amdgcn_wmma_f32_16x16x32_bf16(false, a, false, bf0,
                                                         (short)0, c0, false, false);
            c1 = __builtin_amdgcn_wmma_f32_16x16x32_bf16(false, a, false, bf1,
                                                         (short)0, c1, false, false);
        }

        // ---- online softmax (row stats replicated per 16-lane half) ------
        float p0[8], p1[8];
#pragma unroll
        for (int r = 0; r < 8; ++r) {
            float t = fmaxf(c0[r], c1[r]);
            t = fmaxf(t, __shfl_xor(t, 1, 32));
            t = fmaxf(t, __shfl_xor(t, 2, 32));
            t = fmaxf(t, __shfl_xor(t, 4, 32));
            t = fmaxf(t, __shfl_xor(t, 8, 32));
            const float nm = fmaxf(mrow[r], t);
            const float sc = __expf(mrow[r] - nm);
            mrow[r] = nm;
            p0[r] = __expf(c0[r] - nm);
            p1[r] = __expf(c1[r] - nm);
            float rs = p0[r] + p1[r];
            rs += __shfl_xor(rs, 1, 32);
            rs += __shfl_xor(rs, 2, 32);
            rs += __shfl_xor(rs, 4, 32);
            rs += __shfl_xor(rs, 8, 32);
            lrow[r] = lrow[r] * sc + rs;
#pragma unroll
            for (int n = 0; n < 8; ++n) acc[n][r] *= sc;
        }

        // ---- transpose P (C-layout) -> A-layout via per-wave LDS ---------
#pragma unroll
        for (int r = 0; r < 8; ++r) {
            const int prow = half * 8 + r;
            myP[prow * PSTRIDE + l16]      = (__bf16)p0[r];
            myP[prow * PSTRIDE + 16 + l16] = (__bf16)p1[r];
        }
        asm volatile("" ::: "memory");  // keep LDS stores before the reloads
        const v16bf pa = join16(*(const v8bf*)(myP + l16 * PSTRIDE + half * 16),
                                *(const v8bf*)(myP + l16 * PSTRIDE + half * 16 + 8));

        // ---- out_tile += P @ V  (B[k][d] = Vt[dbase+d][kb+k], contiguous) -
#pragma unroll
        for (int n = 0; n < 8; ++n) {
            const __bf16* vp = vmat + (size_t)(dbase + n * 16 + l16) * SS + kb + half * 16;
            __builtin_prefetch((const void*)(vp + KTILE), 0, 0);  // next key tile
            const v16bf bf = join16(*(const v8bf*)vp, *(const v8bf*)(vp + 8));
            acc[n] = __builtin_amdgcn_wmma_f32_16x16x32_bf16(false, pa, false, bf,
                                                             (short)0, acc[n], false, false);
        }
        __syncthreads();   // everyone done with ldsK[buf] before it is refilled
    }

    // ---- epilogue: normalize by l and store f32 --------------------------
#pragma unroll
    for (int r = 0; r < 8; ++r) {
        const float inv = 1.0f / lrow[r];
        const int   row = qbase + w * 16 + half * 8 + r;
#pragma unroll
        for (int n = 0; n < 8; ++n) {
            out[((size_t)b * SS + row) * EE + dbase + n * 16 + l16] = acc[n][r] * inv;
        }
    }
}

// ==========================================================================
// Host launcher
// ==========================================================================
extern "C" void kernel_launch(void* const* d_in, const int* in_sizes, int n_in,
                              void* d_out, int out_size, void* d_ws, size_t ws_size,
                              hipStream_t stream)
{
    (void)in_sizes; (void)n_in; (void)out_size; (void)ws_size;

    const float* q  = (const float*)d_in[0];
    const float* k  = (const float*)d_in[1];
    const float* v  = (const float*)d_in[2];
    const float* Wq = (const float*)d_in[3];
    const float* bq = (const float*)d_in[4];
    const float* Wk = (const float*)d_in[5];
    const float* bk = (const float*)d_in[6];
    const float* Wv = (const float*)d_in[7];
    const float* bv = (const float*)d_in[8];

    const size_t nElem = (size_t)BB * SS * EE;          // 8.39M elems, 16 MB bf16
    __bf16* Qb = (__bf16*)d_ws;
    __bf16* Kb = Qb + nElem;
    __bf16* Vt = Kb + nElem;

    proj_kernel<<<dim3(EE / 16, (BB * SS) / 64, 3), 128, 0, stream>>>(
        q, k, v, Wq, bq, Wk, bk, Wv, bv, Qb, Kb, Vt);

    flash_kernel<<<dim3(EE / DCHUNK, SS / QTILE, BB), 128, 0, stream>>>(
        Qb, Kb, Vt, (float*)d_out);
}
